// MultiHeadAttention_39822936768790
// MI455X (gfx1250) — matrix operand order repaired and checked by reading, not compiled, not run on hardware
//
#include <hip/hip_runtime.h>

#define BB   2
#define SS   2048
#define DDIM 2048
#define HH   16
#define DKK  128

typedef __attribute__((ext_vector_type(16))) _Float16 v16h;
typedef __attribute__((ext_vector_type(8)))  float    v8f;
typedef unsigned int u32x4 __attribute__((ext_vector_type(4)));
typedef int          i32x4 __attribute__((ext_vector_type(4)));
typedef int          i32x8 __attribute__((ext_vector_type(8)));

#if defined(__gfx1250__) && __has_builtin(__builtin_amdgcn_tensor_load_to_lds) && \
    __has_builtin(__builtin_amdgcn_s_wait_tensorcnt)
#define USE_TDM 1
#else
#define USE_TDM 0
#endif

__device__ __forceinline__ v8f wmma16(v16h a, v16h b, v8f c) {
  // D = A(16x32 f16) * B(32x16 f16) + C(16x16 f32)
  return __builtin_amdgcn_wmma_f32_16x16x32_f16(
      /*neg_a=*/false, a, /*neg_b=*/false, b,
      /*c_mod=*/(short)0, c, /*reuse_a=*/false, /*reuse_b=*/false);
}

// ---------------------------------------------------------------------------
// f32 -> f16 elementwise conversion (for weights)
// ---------------------------------------------------------------------------
__global__ void cvt_f32_f16_kernel(const float* __restrict__ in,
                                   _Float16* __restrict__ out, size_t n) {
  size_t i = (size_t)blockIdx.x * blockDim.x + threadIdx.x;
  if (i < n) out[i] = (_Float16)in[i];
}

// ---------------------------------------------------------------------------
// GEMM: C[M,N] = A[M,K] * W[N,K]^T + bias[N]
//   Block tile 128(M) x 128(N); 8 waves, each 16(M) x 128(N).
//   W k-slab (128 rows x 32 halves) staged in LDS, double-buffered, fetched by
//   the Tensor Data Mover (TDM pads rows 64B->80B for bank-conflict relief).
//   A: f32 (A_F16=false) or f16 (A_F16=true), row-major [M,K].
//   C: f16 (OUT_F16) else f32. STORE_VT scatters into per-head V^T [B,H,dk,S].
// ---------------------------------------------------------------------------
#define WROWS 40                         // padded LDS row stride in halves (80 B)

template <bool A_F16, bool OUT_F16, bool STORE_VT>
__global__ __launch_bounds__(256) void gemm_bias_kernel(
    const void* __restrict__ Ap, const _Float16* __restrict__ Wh,
    const float* __restrict__ bias, void* __restrict__ Cp,
    int M, int N, int K) {
  __shared__ _Float16 lw[2][128 * WROWS];  // 2 x 10240 B
  const int tid  = threadIdx.x;
  const int lane = tid & 31;
  const int wave = tid >> 5;
  const int mrow = lane & 15;
  const int aoff = (lane < 16) ? 0 : 16;  // A fragment K-offset (contiguous, matches B)
  const int boff = (lane < 16) ? 0 : 16;  // B fragment K-offset (contiguous, matches A)
  const int row0 = blockIdx.y * 128 + wave * 16;
  const int col0 = blockIdx.x * 128;

  auto stage = [&](int buf, int kc) {
#if USE_TDM
    if (wave == 0) {
      unsigned long long ga =
          (unsigned long long)(Wh + (size_t)col0 * K + kc);  // tile start (bytes)
      unsigned ldso = (unsigned)(unsigned long long)(
          (__attribute__((address_space(3))) _Float16*)(&lw[buf][0]));
      u32x4 g0;
      g0[0] = 1u;                                   // count=1 (valid user D#)
      g0[1] = ldso;                                 // lds_addr (bytes)
      g0[2] = (unsigned)ga;                         // global_addr[31:0]
      g0[3] = (unsigned)((ga >> 32) & 0x1FFFFFFull) | (2u << 30);  // addr[56:32]|type=2
      i32x8 g1;
      g1[0] = (1 << 16)                 // data_size = 2 bytes
            | (1 << 20)                 // pad_enable
            | (3 << 22)                 // pad_interval: every 16 DWORDs (64 B)
            | (3 << 25);                // pad_amount: 4 DWORDs (16 B) -> 80 B rows
      g1[1] = (int)(((unsigned)K & 0xFFFFu) << 16);                     // dim0 lo16
      g1[2] = (int)(((unsigned)K >> 16) | (((unsigned)N & 0xFFFFu) << 16)); // dim0 hi|dim1 lo
      g1[3] = (int)(((unsigned)N >> 16) | (32u << 16));                 // dim1 hi | tile_dim0=32
      g1[4] = 128;                                                      // tile_dim1 = 128 rows
      g1[5] = K;                                                        // dim0_stride (elems)
      g1[6] = 0;
      g1[7] = 0;
      i32x4 z4 = {0, 0, 0, 0};
#if __clang_major__ >= 23
      i32x8 z8 = {0, 0, 0, 0, 0, 0, 0, 0};
      __builtin_amdgcn_tensor_load_to_lds(g0, g1, z4, z4, z8, 0);
#else
      __builtin_amdgcn_tensor_load_to_lds(g0, g1, z4, z4, 0);
#endif
    }
#else
    // cooperative fallback: 128 rows x 64 B, one 16 B segment per (thread,s)
#pragma unroll
    for (int s = 0; s < 2; ++s) {
      int idx = tid + s * 256;
      int row = idx >> 2, seg = idx & 3;
      const uint4* gp = (const uint4*)(Wh + (size_t)(col0 + row) * K + kc + seg * 8);
      uint4 v = *gp;
      *(uint4*)&lw[buf][row * WROWS + seg * 8] = v;
    }
#endif
  };

  v8f acc[8] = {};
  const size_t arow = (size_t)(row0 + mrow) * K;

  stage(0, 0);
#if USE_TDM
  if (wave == 0) __builtin_amdgcn_s_wait_tensorcnt(0);
#endif
  __syncthreads();

  int cur = 0;
  for (int kc = 0; kc < K; kc += 32, cur ^= 1) {
    if (kc + 32 < K) stage(cur ^ 1, kc + 32);  // prefetch next slab
    v16h a;
    if constexpr (A_F16) {
      const _Float16* A = (const _Float16*)Ap + arow + kc + aoff;
#pragma unroll
      for (int i = 0; i < 8; ++i) { a[i] = A[i]; a[i + 8] = A[8 + i]; }
    } else {
      const float* A = (const float*)Ap + arow + kc + aoff;
#pragma unroll
      for (int i = 0; i < 8; ++i) {
        a[i] = (_Float16)A[i];
        a[i + 8] = (_Float16)A[8 + i];
      }
    }
#pragma unroll
    for (int t = 0; t < 8; ++t) {
      const _Float16* wr = &lw[cur][(t * 16 + mrow) * WROWS + boff];
      v16h b;
#pragma unroll
      for (int i = 0; i < 16; ++i) b[i] = wr[i];
      acc[t] = wmma16(a, b, acc[t]);
    }
#if USE_TDM
    if (wave == 0) __builtin_amdgcn_s_wait_tensorcnt(0);
#endif
    __syncthreads();
  }

#pragma unroll
  for (int t = 0; t < 8; ++t) {
#pragma unroll
    for (int j = 0; j < 8; ++j) {
      int r = row0 + j + ((lane < 16) ? 0 : 8);  // C layout: rows 0-7 | 8-15
      int c = col0 + t * 16 + mrow;
      float v = acc[t][j] + bias[c];
      if constexpr (OUT_F16) {
        _Float16* O = (_Float16*)Cp;
        if constexpr (STORE_VT) {
          // (r = b*S+s, c = h*dk+d)  ->  VT[((b*H+h)*dk+d)*S + s]
          int bb = r / SS, s = r % SS, h = c / DKK, d = c % DKK;
          O[(((size_t)bb * HH + h) * DKK + d) * SS + s] = (_Float16)v;
        } else {
          O[(size_t)r * N + c] = (_Float16)v;
        }
      } else {
        ((float*)Cp)[(size_t)r * N + c] = v;
      }
    }
  }
}

// ---------------------------------------------------------------------------
// RoPE on f16 Q and K stored as [B,S,H*dk]. One thread per rotation pair.
// ---------------------------------------------------------------------------
__global__ void rope_kernel(_Float16* __restrict__ Q, _Float16* __restrict__ K) {
  size_t idx = (size_t)blockIdx.x * blockDim.x + threadIdx.x;
  const size_t total = (size_t)BB * SS * HH * (DKK / 2);
  if (idx >= total) return;
  int d2 = idx & 63;
  size_t rem = idx >> 6;
  int h = rem % HH; rem /= HH;
  int s = rem % SS;
  int b = (int)(rem / SS);
  // inv_freq = 10000^{-(2*d2)/128} = 2^{-d2 * log2(10000)/64}
  float inv = exp2f(-(float)d2 * (13.287712379549449f / 64.0f));
  float ang = (float)s * inv;
  float cs = cosf(ang), sn = sinf(ang);
  size_t base = ((size_t)(b * SS + s)) * DDIM + (size_t)h * DKK;
  float q1 = (float)Q[base + d2], q2 = (float)Q[base + d2 + 64];
  Q[base + d2]      = (_Float16)(q1 * cs - q2 * sn);
  Q[base + d2 + 64] = (_Float16)(q2 * cs + q1 * sn);
  float k1 = (float)K[base + d2], k2 = (float)K[base + d2 + 64];
  K[base + d2]      = (_Float16)(k1 * cs - k2 * sn);
  K[base + d2 + 64] = (_Float16)(k2 * cs + k1 * sn);
}

// ---------------------------------------------------------------------------
// Flash attention: per wave, one 16-query tile of one (b,h); 8 waves/block
// share one (b,h) so K/V re-reads hit the WGP cache.
// Q,K: f16 [B,S,H*dk] (post-RoPE). VT: f16 [B,H,dk,S]. X: f16 [B,S,H*dk].
// ---------------------------------------------------------------------------
__global__ __launch_bounds__(256) void flash_attn_kernel(
    const _Float16* __restrict__ Qm, const _Float16* __restrict__ Km,
    const _Float16* __restrict__ Vt, _Float16* __restrict__ Xm) {
  __shared__ _Float16 ps_all[8][16 * 32];  // per-wave P staging (C->A relayout)
  const int lane = threadIdx.x & 31;
  const int wave = threadIdx.x >> 5;
  const int mrow = lane & 15;
  const int aoff = (lane < 16) ? 0 : 16;
  const int boff = (lane < 16) ? 0 : 16;

  const int tiles_per_seq = SS / 16;  // 128
  int tile = blockIdx.x * 8 + wave;
  int q0 = (tile % tiles_per_seq) * 16;
  int bh = tile / tiles_per_seq;
  int h = bh % HH;
  int b = bh / HH;

  const float scale = 0.08838834764831845f;  // 1/sqrt(128), folded into Q

  v16h qa[4];
  {
    const _Float16* Qr = Qm + ((size_t)(b * SS + q0 + mrow)) * DDIM + (size_t)h * DKK;
#pragma unroll
    for (int c = 0; c < 4; ++c) {
      const _Float16* p = Qr + c * 32 + aoff;
#pragma unroll
      for (int i = 0; i < 8; ++i) {
        qa[c][i]     = (_Float16)((float)p[i] * scale);
        qa[c][i + 8] = (_Float16)((float)p[8 + i] * scale);
      }
    }
  }

  v8f o[8] = {};
  float mrun[8], lrun[8];
#pragma unroll
  for (int j = 0; j < 8; ++j) { mrun[j] = -3.0e38f; lrun[j] = 0.0f; }

  _Float16* ps = ps_all[wave];
  const _Float16* Kbase = Km + ((size_t)b * SS) * DDIM + (size_t)h * DKK;
  const _Float16* Vbase = Vt + ((size_t)(b * HH + h)) * DKK * SS;

  for (int kb = 0; kb < SS; kb += 32) {
    // ---- scores: 16 x 32 via 8 WMMAs -------------------------------------
    v8f s0 = {}, s1 = {};
#pragma unroll
    for (int c = 0; c < 4; ++c) {
      const _Float16* K0 = Kbase + (size_t)(kb + mrow) * DDIM + c * 32 + boff;
      const _Float16* K1 = K0 + (size_t)16 * DDIM;
      v16h b0, b1;
#pragma unroll
      for (int i = 0; i < 16; ++i) { b0[i] = K0[i]; b1[i] = K1[i]; }
      s0 = wmma16(qa[c], b0, s0);
      s1 = wmma16(qa[c], b1, s1);
    }
    // ---- online softmax (half-wave reductions) ---------------------------
#pragma unroll
    for (int j = 0; j < 8; ++j) {
      float v = fmaxf(s0[j], s1[j]);
#pragma unroll
      for (int off = 1; off < 16; off <<= 1)
        v = fmaxf(v, __shfl_xor(v, off, 32));
      float mnew = fmaxf(mrun[j], v);
      float alpha = __expf(mrun[j] - mnew);
      float p0 = __expf(s0[j] - mnew);
      float p1 = __expf(s1[j] - mnew);
      float rs = p0 + p1;
#pragma unroll
      for (int off = 1; off < 16; off <<= 1)
        rs += __shfl_xor(rs, off, 32);
      mrun[j] = mnew;
      lrun[j] = lrun[j] * alpha + rs;
      s0[j] = p0;
      s1[j] = p1;
#pragma unroll
      for (int t = 0; t < 8; ++t) o[t][j] *= alpha;
    }
    // ---- stage P (C layout) to LDS as row-major 16x32 --------------------
#pragma unroll
    for (int j = 0; j < 8; ++j) {
      int m = j + ((lane < 16) ? 0 : 8);
      ps[m * 32 + mrow]      = (_Float16)s0[j];
      ps[m * 32 + 16 + mrow] = (_Float16)s1[j];
    }
    __builtin_amdgcn_wave_barrier();
    asm volatile("s_wait_dscnt 0" ::: "memory");
    v16h pa;
    {
      const _Float16* pr = ps + mrow * 32 + aoff;
#pragma unroll
      for (int i = 0; i < 8; ++i) { pa[i] = pr[i]; pa[i + 8] = pr[8 + i]; }
    }
    __builtin_amdgcn_wave_barrier();
    // ---- O += P * V : 8 WMMAs over dk=128 --------------------------------
#pragma unroll
    for (int t = 0; t < 8; ++t) {
      const _Float16* Vr = Vbase + (size_t)(t * 16 + mrow) * SS + kb + boff;
      v16h vb;
#pragma unroll
      for (int i = 0; i < 16; ++i) vb[i] = Vr[i];
      o[t] = wmma16(pa, vb, o[t]);
    }
  }

  float invl[8];
#pragma unroll
  for (int j = 0; j < 8; ++j) invl[j] = 1.0f / lrun[j];
#pragma unroll
  for (int t = 0; t < 8; ++t) {
#pragma unroll
    for (int j = 0; j < 8; ++j) {
      int m = j + ((lane < 16) ? 0 : 8);
      Xm[((size_t)(b * SS + q0 + m)) * DDIM + (size_t)h * DKK + t * 16 + mrow] =
          (_Float16)(o[t][j] * invl[j]);
    }
  }
}

// ---------------------------------------------------------------------------
extern "C" void kernel_launch(void* const* d_in, const int* in_sizes, int n_in,
                              void* d_out, int out_size, void* d_ws, size_t ws_size,
                              hipStream_t stream) {
  const float* query = (const float*)d_in[0];
  const float* key_i = (const float*)d_in[1];
  const float* value = (const float*)d_in[2];
  const float* Wq = (const float*)d_in[3];
  const float* bq = (const float*)d_in[4];
  const float* Wk = (const float*)d_in[5];
  const float* bk = (const float*)d_in[6];
  const float* Wv = (const float*)d_in[7];
  const float* bv = (const float*)d_in[8];
  const float* Wo = (const float*)d_in[9];
  const float* bo = (const float*)d_in[10];
  float* out = (float*)d_out;

  const size_t n = (size_t)BB * SS * DDIM;  // 8.39M elements per activation
  const size_t w = (size_t)DDIM * DDIM;     // 4.19M elements per weight
  _Float16* ws  = (_Float16*)d_ws;          // ~101 MB total
  _Float16* Wqh = ws;
  _Float16* Wkh = Wqh + w;
  _Float16* Wvh = Wkh + w;
  _Float16* Woh = Wvh + w;
  _Float16* Qf  = Woh + w;
  _Float16* Kf  = Qf + n;
  _Float16* VT  = Kf + n;
  _Float16* Xf  = VT + n;

  const int cb = 256;
  cvt_f32_f16_kernel<<<(int)((w + cb - 1) / cb), cb, 0, stream>>>(Wq, Wqh, w);
  cvt_f32_f16_kernel<<<(int)((w + cb - 1) / cb), cb, 0, stream>>>(Wk, Wkh, w);
  cvt_f32_f16_kernel<<<(int)((w + cb - 1) / cb), cb, 0, stream>>>(Wv, Wvh, w);
  cvt_f32_f16_kernel<<<(int)((w + cb - 1) / cb), cb, 0, stream>>>(Wo, Woh, w);

  dim3 gblk(DDIM / 128, (BB * SS) / 128);  // (16, 32)
  gemm_bias_kernel<false, true, false><<<gblk, 256, 0, stream>>>(query, Wqh, bq, Qf, BB * SS, DDIM, DDIM);
  gemm_bias_kernel<false, true, false><<<gblk, 256, 0, stream>>>(key_i, Wkh, bk, Kf, BB * SS, DDIM, DDIM);
  gemm_bias_kernel<false, true, true ><<<gblk, 256, 0, stream>>>(value, Wvh, bv, VT, BB * SS, DDIM, DDIM);

  size_t rope_threads = (size_t)BB * SS * HH * (DKK / 2);
  rope_kernel<<<(int)((rope_threads + 255) / 256), 256, 0, stream>>>(Qf, Kf);

  flash_attn_kernel<<<(BB * HH * SS / 16) / 8, 256, 0, stream>>>(Qf, Kf, VT, Xf);

  gemm_bias_kernel<true, false, false><<<gblk, 256, 0, stream>>>(Xf, Woh, bo, out, BB * SS, DDIM, DDIM);
}
